// CrossAttention_27101243638344
// MI455X (gfx1250) — matrix operand order repaired and checked by reading, not compiled, not run on hardware
//
#include <hip/hip_runtime.h>
#include <hip/hip_bf16.h>
#include <math.h>

#define BATCH 8
#define SQ    2048
#define EMB   1024

typedef __attribute__((ext_vector_type(16))) __bf16   v16bf;
typedef __attribute__((ext_vector_type(8)))  __bf16   v8bf;
typedef __attribute__((ext_vector_type(4)))  __bf16   v4bf;
typedef __attribute__((ext_vector_type(8)))  float    v8f;
typedef __attribute__((ext_vector_type(4)))  unsigned u32x4;
typedef __attribute__((ext_vector_type(8)))  unsigned u32x8;

// ---------------------------------------------------------------------------
// Fragment loaders per CDNA5 WMMA VGPR layouts (CDNA5 ISA §7.12.2).
// LDS row stride 40 bf16 (80 B) keeps every chunk 16B-aligned -> ds_load_b128.
// ---------------------------------------------------------------------------
__device__ inline v16bf load_a_frag(const __bf16* sm, int ld, int mbase, int lane) {
    int m    = mbase + (lane & 15);
    int koff = (lane < 16) ? 0 : 8;
    const __bf16* row = sm + m * ld + koff;
    v8bf lo = *(const v8bf*)(row);
    v8bf hi = *(const v8bf*)(row + 16);
    return __builtin_shufflevector(lo, hi, 0,1,2,3,4,5,6,7,8,9,10,11,12,13,14,15);
}

__device__ inline v16bf load_b_frag_T(const __bf16* sm, int ld, int nbase, int lane) {
    int n    = nbase + (lane & 15);
    int koff = (lane < 16) ? 0 : 8;
    const __bf16* row = sm + n * ld + koff;
    v8bf lo = *(const v8bf*)(row);
    v8bf hi = *(const v8bf*)(row + 16);
    return __builtin_shufflevector(lo, hi, 0,1,2,3,4,5,6,7,8,9,10,11,12,13,14,15);
}

// ---------------------------------------------------------------------------
// Tensor Data Mover: D# groups 0/1, 2D bf16 tile, pad_enable so LDS rows land
// with stride 40 bf16 (pad 4 dwords every 16 dwords).  (CDNA5 ISA)
// ---------------------------------------------------------------------------
__device__ inline void tdm_load_tile_bf16(const __bf16* gptr, unsigned lds_off,
                                          unsigned tensor_d0, unsigned tensor_d1,
                                          unsigned tile_d0, unsigned tile_d1,
                                          unsigned stride0) {
    unsigned long long ga = (unsigned long long)(size_t)gptr;
    u32x4 g0;
    g0[0] = 1u;                                                 // count=1
    g0[1] = lds_off;                                            // lds_addr (bytes)
    g0[2] = (unsigned)(ga & 0xFFFFFFFFu);                       // global_addr[31:0]
    g0[3] = (unsigned)((ga >> 32) & 0x01FFFFFFu) | (2u << 30);  // addr[56:32] | type=2
    u32x8 g1;
    g1[0] = (1u << 16) | (1u << 20) | (3u << 22) | (3u << 25);  // 2B, pad 4dw/16dw
    g1[1] = (tensor_d0 & 0xFFFFu) << 16;
    g1[2] = (tensor_d0 >> 16) | ((tensor_d1 & 0xFFFFu) << 16);
    g1[3] = (tensor_d1 >> 16) | (tile_d0 << 16);
    g1[4] = tile_d1;
    g1[5] = stride0;
    g1[6] = 0u;
    g1[7] = 0u;
    asm volatile("tensor_load_to_lds %0, %1" :: "s"(g0), "s"(g1) : "memory");
}

// ---------------------------------------------------------------------------
// Kernel 1: QKV projection.  Out[b,s,f] = sum_e In[b,s,e]*W[f,e] + bias[f]
// Block tile 128(M=s) x 64(N=f), 8 waves (4x2), 32x32 output per wave.
// ---------------------------------------------------------------------------
__global__ __launch_bounds__(256) void qkv_proj_kernel(
    const float* __restrict__ x, const float* __restrict__ ctx,
    const float* __restrict__ Wq, const float* __restrict__ bq,
    const float* __restrict__ Wk, const float* __restrict__ bk,
    const float* __restrict__ Wv, const float* __restrict__ bv,
    __bf16* __restrict__ Qo, __bf16* __restrict__ Ko, __bf16* __restrict__ Vo)
{
    int bz = blockIdx.z;
    int b = bz / 3, which = bz % 3;
    const float* in   = (which == 0) ? x  : ctx;
    const float* W    = (which == 0) ? Wq : (which == 1 ? Wk : Wv);
    const float* bias = (which == 0) ? bq : (which == 1 ? bk : bv);
    __bf16*      out  = (which == 0) ? Qo : (which == 1 ? Ko : Vo);

    int s0 = blockIdx.x * 128;
    int n0 = blockIdx.y * 64;

    __shared__ __align__(16) __bf16 As[128][40];
    __shared__ __align__(16) __bf16 Bs[64][40];

    int tid  = threadIdx.x;
    int lane = tid & 31;
    int wave = tid >> 5;
    int wm = wave & 3;          // 4 x 32 rows
    int wn = wave >> 2;         // 2 x 32 cols

    v8f acc[2][2] = {};
    const float* inb = in + (size_t)b * SQ * EMB;

    for (int k0 = 0; k0 < EMB; k0 += 32) {
#pragma unroll
        for (int r = 0; r < 4; ++r) {                 // A: 128x32 f32 = 1024 float4
            int idx = r * 256 + tid;
            int mm = idx >> 3, kk = (idx & 7) * 4;
            float4 fa = *(const float4*)&inb[(size_t)(s0 + mm) * EMB + (k0 + kk)];
            v4bf pa = { (__bf16)fa.x, (__bf16)fa.y, (__bf16)fa.z, (__bf16)fa.w };
            *(v4bf*)&As[mm][kk] = pa;
        }
#pragma unroll
        for (int r = 0; r < 2; ++r) {                 // B: 64x32 f32 = 512 float4
            int idx = r * 256 + tid;
            int mm = idx >> 3, kk = (idx & 7) * 4;
            float4 fw = *(const float4*)&W[(size_t)(n0 + mm) * EMB + (k0 + kk)];
            v4bf pw = { (__bf16)fw.x, (__bf16)fw.y, (__bf16)fw.z, (__bf16)fw.w };
            *(v4bf*)&Bs[mm][kk] = pw;
        }
        __syncthreads();
        if (k0 + 32 < EMB) {
            __builtin_prefetch(&inb[(size_t)(s0 + (tid & 127)) * EMB + k0 + 32], 0, 1);
            __builtin_prefetch(&W  [(size_t)(n0 + (tid & 63))  * EMB + k0 + 32], 0, 1);
        }
        v16bf a0 = load_a_frag  (&As[0][0], 40, wm * 32,      lane);
        v16bf a1 = load_a_frag  (&As[0][0], 40, wm * 32 + 16, lane);
        v16bf b0 = load_b_frag_T(&Bs[0][0], 40, wn * 32,      lane);
        v16bf b1 = load_b_frag_T(&Bs[0][0], 40, wn * 32 + 16, lane);
        acc[0][0] = __builtin_amdgcn_wmma_f32_16x16x32_bf16(false, a0, false, b0, (short)0, acc[0][0], false, false);
        acc[0][1] = __builtin_amdgcn_wmma_f32_16x16x32_bf16(false, a0, false, b1, (short)0, acc[0][1], false, false);
        acc[1][0] = __builtin_amdgcn_wmma_f32_16x16x32_bf16(false, a1, false, b0, (short)0, acc[1][0], false, false);
        acc[1][1] = __builtin_amdgcn_wmma_f32_16x16x32_bf16(false, a1, false, b1, (short)0, acc[1][1], false, false);
        __syncthreads();
    }

    int moff = (lane < 16) ? 0 : 8;
    int ncol = lane & 15;
#pragma unroll
    for (int ti = 0; ti < 2; ++ti) {
#pragma unroll
        for (int tj = 0; tj < 2; ++tj) {
            int ng = n0 + wn * 32 + tj * 16 + ncol;
            float bb = bias[ng];
#pragma unroll
            for (int i = 0; i < 8; ++i) {
                int m = s0 + wm * 32 + ti * 16 + i + moff;
                out[((size_t)b * SQ + m) * EMB + ng] = (__bf16)(acc[ti][tj][i] + bb);
            }
        }
    }
}

// ---------------------------------------------------------------------------
// Kernel 2: att_raw[b,i,j] = mask ? -1e9 : scale * sum_d Q[b,i,d]*K[b,j,d]
// TDM-fetched bf16 tiles, double-buffered.  128(i) x 64(j) per block.
// ---------------------------------------------------------------------------
__global__ __launch_bounds__(256) void scores_kernel(
    const __bf16* __restrict__ Q, const __bf16* __restrict__ K,
    const unsigned char* __restrict__ mask, float* __restrict__ att)
{
    int b  = blockIdx.z;
    int i0 = blockIdx.x * 128;
    int j0 = blockIdx.y * 64;

    __shared__ __align__(16) __bf16 Qs[2][128][40];
    __shared__ __align__(16) __bf16 Ks[2][64][40];

    int tid  = threadIdx.x;
    int lane = tid & 31;
    int wave = tid >> 5;
    int wm = wave & 3;
    int wn = wave >> 2;

    v8f acc[2][2] = {};
    const __bf16* Qb = Q + (size_t)b * SQ * EMB;
    const __bf16* Kb = K + (size_t)b * SQ * EMB;

    if (wave == 0) {   // prologue: TDM-fetch chunk 0 into buffer 0
        tdm_load_tile_bf16(Qb + (size_t)i0 * EMB, (unsigned)(size_t)&Qs[0][0][0],
                           EMB, SQ, 32, 128, EMB);
        tdm_load_tile_bf16(Kb + (size_t)j0 * EMB, (unsigned)(size_t)&Ks[0][0][0],
                           EMB, SQ, 32, 64, EMB);
    }

    int cur = 0;
    for (int k0 = 0; k0 < EMB; k0 += 32) {
        if (wave == 0) {
            if (k0 + 32 < EMB) {
                tdm_load_tile_bf16(Qb + (size_t)i0 * EMB + (k0 + 32),
                                   (unsigned)(size_t)&Qs[cur ^ 1][0][0],
                                   EMB, SQ, 32, 128, EMB);
                tdm_load_tile_bf16(Kb + (size_t)j0 * EMB + (k0 + 32),
                                   (unsigned)(size_t)&Ks[cur ^ 1][0][0],
                                   EMB, SQ, 32, 64, EMB);
                __builtin_amdgcn_s_wait_tensorcnt(2);   // chunk k0 retired
            } else {
                __builtin_amdgcn_s_wait_tensorcnt(0);
            }
        }
        __syncthreads();

        v16bf a0 = load_a_frag  (&Qs[cur][0][0], 40, wm * 32,      lane);
        v16bf a1 = load_a_frag  (&Qs[cur][0][0], 40, wm * 32 + 16, lane);
        v16bf b0 = load_b_frag_T(&Ks[cur][0][0], 40, wn * 32,      lane);
        v16bf b1 = load_b_frag_T(&Ks[cur][0][0], 40, wn * 32 + 16, lane);
        acc[0][0] = __builtin_amdgcn_wmma_f32_16x16x32_bf16(false, a0, false, b0, (short)0, acc[0][0], false, false);
        acc[0][1] = __builtin_amdgcn_wmma_f32_16x16x32_bf16(false, a0, false, b1, (short)0, acc[0][1], false, false);
        acc[1][0] = __builtin_amdgcn_wmma_f32_16x16x32_bf16(false, a1, false, b0, (short)0, acc[1][0], false, false);
        acc[1][1] = __builtin_amdgcn_wmma_f32_16x16x32_bf16(false, a1, false, b1, (short)0, acc[1][1], false, false);
        __syncthreads();
        cur ^= 1;
    }

    const float scale = 0.03125f;   // 1/sqrt(1024)
    int moff = (lane < 16) ? 0 : 8;
    int ncol = lane & 15;
#pragma unroll
    for (int ti = 0; ti < 2; ++ti) {
#pragma unroll
        for (int tj = 0; tj < 2; ++tj) {
            int jg = j0 + wn * 32 + tj * 16 + ncol;
#pragma unroll
            for (int i = 0; i < 8; ++i) {
                int ig = i0 + wm * 32 + ti * 16 + i + moff;
                size_t idx = ((size_t)b * SQ + ig) * SQ + jg;
                att[idx] = mask[idx] ? -1e9f : (scale * acc[ti][tj][i]);
            }
        }
    }
}

// ---------------------------------------------------------------------------
// Kernel 3: in-place softmax over last axis (row length SQ), one block/row.
// ---------------------------------------------------------------------------
__global__ __launch_bounds__(256) void softmax_kernel(float* __restrict__ att)
{
    size_t row = blockIdx.x;
    float* p = att + row * SQ;
    int tid = threadIdx.x;

    __shared__ float red[256];

    float vals[8];
    float lmax = -INFINITY;
#pragma unroll
    for (int r = 0; r < 8; ++r) {
        vals[r] = p[tid + r * 256];
        lmax = fmaxf(lmax, vals[r]);
    }
    red[tid] = lmax;
    __syncthreads();
    for (int s = 128; s > 0; s >>= 1) {
        if (tid < s) red[tid] = fmaxf(red[tid], red[tid + s]);
        __syncthreads();
    }
    float m = red[0];
    __syncthreads();

    float lsum = 0.f;
#pragma unroll
    for (int r = 0; r < 8; ++r) {
        vals[r] = __expf(vals[r] - m);
        lsum += vals[r];
    }
    red[tid] = lsum;
    __syncthreads();
    for (int s = 128; s > 0; s >>= 1) {
        if (tid < s) red[tid] += red[tid + s];
        __syncthreads();
    }
    float inv = 1.f / red[0];
#pragma unroll
    for (int r = 0; r < 8; ++r) p[tid + r * 256] = vals[r] * inv;
}

// ---------------------------------------------------------------------------
// Kernel 4: out[b,i,k] = sum_j att[b,i,j] * V[b,j,k]
// att f32 -> bf16 in LDS; V staged transposed ([n][j]) for contiguous B-frags.
// Block tile 128(i) x 64(k), 32x32 per wave.
// ---------------------------------------------------------------------------
__global__ __launch_bounds__(256) void av_kernel(
    const float* __restrict__ att, const __bf16* __restrict__ V,
    float* __restrict__ out)
{
    int b  = blockIdx.z;
    int i0 = blockIdx.x * 128;
    int n0 = blockIdx.y * 64;

    __shared__ __align__(16) __bf16 As[128][40];  // [i][j-chunk]
    __shared__ __align__(16) __bf16 Vs[64][40];   // [n][j-chunk] (transposed)

    int tid  = threadIdx.x;
    int lane = tid & 31;
    int wave = tid >> 5;
    int wm = wave & 3;
    int wn = wave >> 2;

    v8f acc[2][2] = {};
    const float*  Ab = att + (size_t)b * SQ * SQ;
    const __bf16* Vb = V   + (size_t)b * SQ * EMB;

    for (int j0 = 0; j0 < SQ; j0 += 32) {
#pragma unroll
        for (int r = 0; r < 4; ++r) {                 // att tile 128x32 f32
            int idx = r * 256 + tid;
            int mm = idx >> 3, kk = (idx & 7) * 4;
            float4 fa = *(const float4*)&Ab[(size_t)(i0 + mm) * SQ + (j0 + kk)];
            v4bf pa = { (__bf16)fa.x, (__bf16)fa.y, (__bf16)fa.z, (__bf16)fa.w };
            *(v4bf*)&As[mm][kk] = pa;
        }
#pragma unroll
        for (int r = 0; r < 2; ++r) {                 // V tile 32x64 bf16, transpose
            int idx = r * 256 + tid;
            int jj = idx >> 4, nn = (idx & 15) * 4;
            v4bf pv = *(const v4bf*)&Vb[(size_t)(j0 + jj) * EMB + (n0 + nn)];
            Vs[nn + 0][jj] = pv[0];
            Vs[nn + 1][jj] = pv[1];
            Vs[nn + 2][jj] = pv[2];
            Vs[nn + 3][jj] = pv[3];
        }
        __syncthreads();
        if (j0 + 32 < SQ) {
            __builtin_prefetch(&Ab[(size_t)(i0 + (tid & 127)) * SQ + j0 + 32], 0, 1);
            __builtin_prefetch(&Vb[(size_t)(j0 + 32 + (tid & 31)) * EMB + n0], 0, 1);
        }
        v16bf a0 = load_a_frag  (&As[0][0], 40, wm * 32,      lane);
        v16bf a1 = load_a_frag  (&As[0][0], 40, wm * 32 + 16, lane);
        v16bf b0 = load_b_frag_T(&Vs[0][0], 40, wn * 32,      lane);
        v16bf b1 = load_b_frag_T(&Vs[0][0], 40, wn * 32 + 16, lane);
        acc[0][0] = __builtin_amdgcn_wmma_f32_16x16x32_bf16(false, a0, false, b0, (short)0, acc[0][0], false, false);
        acc[0][1] = __builtin_amdgcn_wmma_f32_16x16x32_bf16(false, a0, false, b1, (short)0, acc[0][1], false, false);
        acc[1][0] = __builtin_amdgcn_wmma_f32_16x16x32_bf16(false, a1, false, b0, (short)0, acc[1][0], false, false);
        acc[1][1] = __builtin_amdgcn_wmma_f32_16x16x32_bf16(false, a1, false, b1, (short)0, acc[1][1], false, false);
        __syncthreads();
    }

    int moff = (lane < 16) ? 0 : 8;
    int ncol = lane & 15;
#pragma unroll
    for (int ti = 0; ti < 2; ++ti) {
#pragma unroll
        for (int tj = 0; tj < 2; ++tj) {
            int ng = n0 + wn * 32 + tj * 16 + ncol;
#pragma unroll
            for (int i = 0; i < 8; ++i) {
                int ig = i0 + wm * 32 + ti * 16 + i + moff;
                out[((size_t)b * SQ + ig) * EMB + ng] = acc[ti][tj][i];
            }
        }
    }
}

// ---------------------------------------------------------------------------
// Launch
// ---------------------------------------------------------------------------
extern "C" void kernel_launch(void* const* d_in, const int* in_sizes, int n_in,
                              void* d_out, int out_size, void* d_ws, size_t ws_size,
                              hipStream_t stream) {
    const float*         x    = (const float*)d_in[0];
    const float*         ctx  = (const float*)d_in[1];
    const unsigned char* mask = (const unsigned char*)d_in[2];  // jnp bool: 1 byte/elem
    const float* Wq = (const float*)d_in[3];
    const float* bq = (const float*)d_in[4];
    const float* Wk = (const float*)d_in[5];
    const float* bk = (const float*)d_in[6];
    const float* Wv = (const float*)d_in[7];
    const float* bv = (const float*)d_in[8];

    float* out = (float*)d_out;                               // (B,S,E)
    float* att = out + (size_t)BATCH * SQ * EMB;              // (B,S,S)

    __bf16* Qw = (__bf16*)d_ws;                               // bf16 scratch: 3 x B*S*E
    __bf16* Kw = Qw + (size_t)BATCH * SQ * EMB;
    __bf16* Vw = Kw + (size_t)BATCH * SQ * EMB;

    dim3 blk(256);

    dim3 g1(SQ / 128, EMB / 64, BATCH * 3);
    qkv_proj_kernel<<<g1, blk, 0, stream>>>(x, ctx, Wq, bq, Wk, bk, Wv, bv, Qw, Kw, Vw);

    dim3 g2(SQ / 128, SQ / 64, BATCH);
    scores_kernel<<<g2, blk, 0, stream>>>(Qw, Kw, mask, att);

    dim3 g3(BATCH * SQ);
    softmax_kernel<<<g3, blk, 0, stream>>>(att);

    dim3 g4(SQ / 128, EMB / 64, BATCH);
    av_kernel<<<g4, blk, 0, stream>>>(att, Vw, out);
}
